// GATv2ConvEncoder_10926396801588
// MI455X (gfx1250) — matrix operand order repaired and checked by reading, not compiled, not run on hardware
//
#include <hip/hip_runtime.h>

#define NHEADS 4
#define HD 64          // per-head dim
#define XDIM 128       // concat(h, h0)
#define FSDIM 256      // H*D
#define NLAYERS 4
#define SLOPE 0.2f

typedef __attribute__((ext_vector_type(8)))  __bf16 v8bf;
typedef __attribute__((ext_vector_type(16))) __bf16 v16bf;
typedef __attribute__((ext_vector_type(8)))  float  v8f;

// ---------- bf16 split-precision helpers (round-to-nearest-even) ----------
__device__ inline unsigned short bf_bits_rne(float x) {
    unsigned u = __float_as_uint(x);
    unsigned r = u + 0x7FFFu + ((u >> 16) & 1u);
    return (unsigned short)(r >> 16);
}
__device__ inline __bf16 us_to_bf(unsigned short s) {
    union { unsigned short u; __bf16 b; } cv; cv.u = s; return cv.b;
}
__device__ inline void split_bf(float x, __bf16& hi, __bf16& lo) {
    unsigned short hb = bf_bits_rne(x);
    float fh = __uint_as_float(((unsigned)hb) << 16);
    hi = us_to_bf(hb);
    lo = us_to_bf(bf_bits_rne(x - fh));
}

// ---------- float atomic max via sign-aware integer atomics ----------
__device__ inline void atomicMaxF(float* addr, float v) {
    if (v >= 0.0f) atomicMax((int*)addr, __float_as_int(v));
    else           atomicMin((unsigned int*)addr, __float_as_uint(v));
}

// ---------- generic zero ----------
__global__ void k_zero(float* p, long long n) {
    long long t = (long long)blockIdx.x * blockDim.x + threadIdx.x;
    if (t < n) p[t] = 0.0f;
}

// ---------- weight prep: W[128,256] f32 -> WT[256][128] bf16 hi/lo ----------
__global__ void k_prep_wt(const float* Ws, const float* Wd,
                          __bf16* WsTh, __bf16* WsTl, __bf16* WdTh, __bf16* WdTl) {
    int t = blockIdx.x * blockDim.x + threadIdx.x;
    if (t >= 2 * FSDIM * XDIM) return;
    int sel = (t >= FSDIM * XDIM);
    int u = sel ? t - FSDIM * XDIM : t;
    int n = u >> 7;      // 0..255 (output col)
    int k = u & 127;     // 0..127 (reduction dim)
    float w = (sel ? Wd : Ws)[k * FSDIM + n];
    __bf16 hi, lo; split_bf(w, hi, lo);
    (sel ? WdTh : WsTh)[u] = hi;
    (sel ? WdTl : WsTl)[u] = lo;
}

// ---------- h0 = feat @ W_in + b_in   [N,16]x[16,64] (tiny K, scalar f32) ----------
__global__ void k_h0(const float* feat, const float* W_in, const float* b_in,
                     float* h0, int N) {
    int t = blockIdx.x * blockDim.x + threadIdx.x;
    if (t >= N * HD) return;
    int i = t >> 6, d = t & 63;
    float s = b_in[d];
    const float* fr = feat + (size_t)i * 16;
#pragma unroll
    for (int k = 0; k < 16; ++k) s += fr[k] * W_in[k * HD + d];
    h0[t] = s;
}

// ---------- x = concat(h, h0) -> split bf16 planes ----------
__global__ void k_build_x(const float* hcur, const float* h0,
                          __bf16* xh, __bf16* xl, int N) {
    int t = blockIdx.x * blockDim.x + threadIdx.x;
    if (t >= N * XDIM) return;
    int i = t >> 7, k = t & 127;
    float v = (k < HD) ? hcur[(size_t)i * HD + k] : h0[(size_t)i * HD + (k - HD)];
    __bf16 hi, lo; split_bf(v, hi, lo);
    xh[t] = hi; xl[t] = lo;
}

// ---------- assemble v16bf from two contiguous 8-element chunks ----------
__device__ inline v16bf ld16(const __bf16* p, int off0, int off1) {
    v8bf lo = *(const v8bf*)(p + off0);
    v8bf hi = *(const v8bf*)(p + off1);
    v16bf r;
#pragma unroll
    for (int i = 0; i < 8; ++i) { r[i] = lo[i]; r[i + 8] = hi[i]; }
    return r;
}

// ---------- WMMA GEMM: fs = x@W_src + b_src, fd = x@W_dst + b_dst ----------
// Split-precision: acc += Ah*Bh + Ah*Bl + Al*Bh  (f32 accumulate)
// Block = 4 waves; each wave owns one 16x16 output tile. ntile 0..15 -> fs, 16..31 -> fd.
__global__ __launch_bounds__(128)
void k_gemm(const __bf16* xh, const __bf16* xl,
            const __bf16* WsTh, const __bf16* WsTl,
            const __bf16* WdTh, const __bf16* WdTl,
            const float* b_src, const float* b_dst,
            float* fs, float* fd) {
    const int lane = threadIdx.x & 31;
    const int wave = threadIdx.x >> 5;
    const int ntile = blockIdx.y * 4 + wave;          // 0..31
    const int mrow = blockIdx.x * 16;
    const int hf = lane >> 4;                          // lane half
    const int mn = lane & 15;                          // A: row m ; B: col n

    const bool isSrc = (ntile < 16);
    const int col = (isSrc ? ntile : ntile - 16) * 16;
    const __bf16* WTh = isSrc ? WsTh : WdTh;
    const __bf16* WTl = isSrc ? WsTl : WdTl;
    const float* bias = isSrc ? b_src : b_dst;
    float* out = isSrc ? fs : fd;

    const __bf16* arow_h = xh + (size_t)(mrow + mn) * XDIM;
    const __bf16* arow_l = xl + (size_t)(mrow + mn) * XDIM;
    const __bf16* bcol_h = WTh + (size_t)(col + mn) * XDIM;
    const __bf16* bcol_l = WTl + (size_t)(col + mn) * XDIM;

    v8f acc = {};
#pragma unroll
    for (int kb = 0; kb < XDIM; kb += 32) {
        // 16-bit A 16x32: lane holds K-runs [kb+8*hf, +8) and [kb+16+8*hf, +8)
        v16bf Ah = ld16(arow_h, kb + 8 * hf, kb + 16 + 8 * hf);
        v16bf Al = ld16(arow_l, kb + 8 * hf, kb + 16 + 8 * hf);
        // 16-bit B 32x16: lane holds column n, K-runs [kb+8*hf, +8) and [kb+16+8*hf, +8) in WT
        v16bf Bh = ld16(bcol_h, kb + 8 * hf, kb + 16 + 8 * hf);
        v16bf Bl = ld16(bcol_l, kb + 8 * hf, kb + 16 + 8 * hf);
        acc = __builtin_amdgcn_wmma_f32_16x16x32_bf16(false, Ah, false, Bh, (short)0, acc, false, false);
        acc = __builtin_amdgcn_wmma_f32_16x16x32_bf16(false, Ah, false, Bl, (short)0, acc, false, false);
        acc = __builtin_amdgcn_wmma_f32_16x16x32_bf16(false, Al, false, Bh, (short)0, acc, false, false);
    }
    float bb = bias[col + mn];
#pragma unroll
    for (int r = 0; r < 8; ++r) {
        // C/D layout: VGPR r -> row (r + 8*hf), col mn
        out[(size_t)(mrow + 8 * hf + r) * FSDIM + col + mn] = acc[r] + bb;
    }
}

// ---------- init per-node softmax state ----------
__global__ void k_init_nd(float* nmax, float* denom, int N) {
    int t = blockIdx.x * blockDim.x + threadIdx.x;
    if (t >= N * NHEADS) return;
    nmax[t] = __uint_as_float(0xFF800000u); // -inf
    denom[t] = 0.0f;
}

// ---------- edge pass A: logits + atomic segment-max ----------
__global__ void k_edge_logits(const float* fs, const float* fd,
                              const int* src, const int* dst, const float* attn,
                              float* logits, float* nmax, int E) {
    int t = blockIdx.x * blockDim.x + threadIdx.x;
    if (t >= E * NHEADS) return;
    int e = t >> 2, h = t & 3;
    int s = src[e], d = dst[e];
    const float4* ps = (const float4*)(fs + (size_t)s * FSDIM + h * HD);
    const float4* pd = (const float4*)(fd + (size_t)d * FSDIM + h * HD);
    const float4* pa = (const float4*)(attn + h * HD);
    float acc = 0.0f;
#pragma unroll
    for (int i = 0; i < 16; ++i) {
        float4 a = ps[i], b = pd[i], w = pa[i];
        float x0 = a.x + b.x, x1 = a.y + b.y, x2 = a.z + b.z, x3 = a.w + b.w;
        x0 = x0 > 0.0f ? x0 : SLOPE * x0;
        x1 = x1 > 0.0f ? x1 : SLOPE * x1;
        x2 = x2 > 0.0f ? x2 : SLOPE * x2;
        x3 = x3 > 0.0f ? x3 : SLOPE * x3;
        acc += w.x * x0 + w.y * x1 + w.z * x2 + w.w * x3;
    }
    logits[t] = acc;
    atomicMaxF(&nmax[(size_t)d * NHEADS + h], acc);
}

// ---------- edge pass B: ex = exp(logit - max); denom += ex; rst += ex*fs[src] ----------
__global__ void k_edge_accum(const float* fs, const int* src, const int* dst,
                             const float* logits, const float* nmax,
                             float* denom, float* rst, int E) {
    int t = blockIdx.x * blockDim.x + threadIdx.x;
    if (t >= E * NHEADS) return;
    int e = t >> 2, h = t & 3;
    int s = src[e], d = dst[e];
    float ex = __expf(logits[t] - nmax[(size_t)d * NHEADS + h]);
    atomicAdd(&denom[(size_t)d * NHEADS + h], ex);
    const float* pf = fs + (size_t)s * FSDIM + h * HD;
    float* pr = rst + (size_t)d * FSDIM + h * HD;
#pragma unroll 8
    for (int i = 0; i < HD; ++i) atomicAdd(&pr[i], ex * pf[i]);
}

// ---------- node stage: h[i,d] = sum_h tanh(rst/denom); isolated nodes -> 0 ----------
__global__ void k_node_out(const float* rst, const float* denom, float* h, int N) {
    int t = blockIdx.x * blockDim.x + threadIdx.x;
    if (t >= N * HD) return;
    int i = t >> 6, d = t & 63;
    float s = 0.0f;
#pragma unroll
    for (int hh = 0; hh < NHEADS; ++hh) {
        float dn = denom[(size_t)i * NHEADS + hh];
        if (dn > 0.0f)
            s += tanhf(rst[(size_t)i * FSDIM + hh * HD + d] / dn);
    }
    h[t] = s;
}

// ---------- readout: hg[g] += is_root[i]*h[i] ----------
__global__ void k_root(const float* h, const float* is_root, const int* gid,
                       float* hg, int N) {
    int t = blockIdx.x * blockDim.x + threadIdx.x;
    if (t >= N * HD) return;
    int i = t >> 6, d = t & 63;
    atomicAdd(&hg[(size_t)gid[i] * HD + d], is_root[i] * h[t]);
}

// ---------- out = hg @ W_out + b_out   [16,64]x[64,32] ----------
__global__ void k_final(const float* hg, const float* Wout, const float* bout, float* out) {
    int t = threadIdx.x;           // 512 = 16*32
    int b = t >> 5, j = t & 31;
    float s = bout[j];
#pragma unroll
    for (int d = 0; d < HD; ++d) s += hg[b * HD + d] * Wout[d * 32 + j];
    out[t] = s;
}

extern "C" void kernel_launch(void* const* d_in, const int* in_sizes, int n_in,
                              void* d_out, int out_size, void* d_ws, size_t ws_size,
                              hipStream_t stream) {
    (void)n_in; (void)out_size; (void)ws_size;
    const float* feat   = (const float*)d_in[0];
    const float* isroot = (const float*)d_in[1];
    const int*   src    = (const int*)d_in[2];
    const int*   dst    = (const int*)d_in[3];
    const int*   gid    = (const int*)d_in[4];
    const float* W_in   = (const float*)d_in[5];
    const float* b_in   = (const float*)d_in[6];
    const float* W_src  = (const float*)d_in[7];
    const float* b_src  = (const float*)d_in[8];
    const float* W_dst  = (const float*)d_in[9];
    const float* b_dst  = (const float*)d_in[10];
    const float* attn   = (const float*)d_in[11];
    const float* W_out  = (const float*)d_in[12];
    const float* b_out  = (const float*)d_in[13];

    const int N = in_sizes[1];   // is_root has N elements (N = 50000, multiple of 16)
    const int E = in_sizes[2];

    // ---- workspace carve-out (256B aligned) ----
    char* base = (char*)d_ws;
    size_t off = 0;
    auto take = [&](size_t bytes) -> void* {
        void* p = base + off;
        off = (off + bytes + 255) & ~(size_t)255;
        return p;
    };
    float*  h0     = (float*) take((size_t)N * HD * 4);
    float*  hbuf   = (float*) take((size_t)N * HD * 4);
    __bf16* xh     = (__bf16*)take((size_t)N * XDIM * 2);
    __bf16* xl     = (__bf16*)take((size_t)N * XDIM * 2);
    float*  fs     = (float*) take((size_t)N * FSDIM * 4);
    float*  fd     = (float*) take((size_t)N * FSDIM * 4); // reused as rst after edge pass A
    float*  logits = (float*) take((size_t)E * NHEADS * 4);
    float*  nmax   = (float*) take((size_t)N * NHEADS * 4);
    float*  denom  = (float*) take((size_t)N * NHEADS * 4);
    __bf16* WsTh   = (__bf16*)take((size_t)FSDIM * XDIM * 2);
    __bf16* WsTl   = (__bf16*)take((size_t)FSDIM * XDIM * 2);
    __bf16* WdTh   = (__bf16*)take((size_t)FSDIM * XDIM * 2);
    __bf16* WdTl   = (__bf16*)take((size_t)FSDIM * XDIM * 2);
    float*  hg     = (float*) take((size_t)16 * HD * 4);

    auto nb = [](long long n, int b) { return (unsigned)((n + b - 1) / b); };

    // weight prep + input projection
    k_prep_wt<<<nb(2LL * FSDIM * XDIM, 256), 256, 0, stream>>>(W_src, W_dst, WsTh, WsTl, WdTh, WdTl);
    k_h0<<<nb((long long)N * HD, 256), 256, 0, stream>>>(feat, W_in, b_in, h0, N);

    const float* hcur = h0;
    for (int layer = 0; layer < NLAYERS; ++layer) {
        k_init_nd<<<nb((long long)N * NHEADS, 256), 256, 0, stream>>>(nmax, denom, N);
        k_build_x<<<nb((long long)N * XDIM, 256), 256, 0, stream>>>(hcur, h0, xh, xl, N);
        k_gemm<<<dim3(N / 16, 8), 128, 0, stream>>>(xh, xl, WsTh, WsTl, WdTh, WdTl,
                                                    b_src, b_dst, fs, fd);
        k_edge_logits<<<nb((long long)E * NHEADS, 256), 256, 0, stream>>>(fs, fd, src, dst, attn,
                                                                          logits, nmax, E);
        // fd is dead now; reuse it as rst accumulator
        k_zero<<<nb((long long)N * FSDIM, 256), 256, 0, stream>>>(fd, (long long)N * FSDIM);
        k_edge_accum<<<nb((long long)E * NHEADS, 256), 256, 0, stream>>>(fs, src, dst, logits,
                                                                         nmax, denom, fd, E);
        k_node_out<<<nb((long long)N * HD, 256), 256, 0, stream>>>(fd, denom, hbuf, N);
        hcur = hbuf;
    }

    k_zero<<<nb(16LL * HD, 256), 256, 0, stream>>>(hg, 16LL * HD);
    k_root<<<nb((long long)N * HD, 256), 256, 0, stream>>>(hbuf, isroot, gid, hg, N);
    k_final<<<1, 512, 0, stream>>>(hg, W_out, b_out, (float*)d_out);
}
